// GIN_Net2_31439160607359
// MI455X (gfx1250) — matrix operand order repaired and checked by reading, not compiled, not run on hardware
//
#include <hip/hip_runtime.h>

#define N_NODES_C 50000
#define IN_F_C    64
#define HIDDEN_C  512
#define N_EDGES_C 160000
#define N_TRAIN_C 80000
#define N_CLASS_C 7

typedef __bf16 bf16;
typedef __attribute__((ext_vector_type(16))) __bf16 v16bf;
typedef __attribute__((ext_vector_type(8)))  __bf16 v8bf;
typedef __attribute__((ext_vector_type(4)))  __bf16 v4bf;
typedef __attribute__((ext_vector_type(8)))  float  v8f;

union V16 { v16bf v; v8bf h[2]; };

// -------------------------------------------------------------------------
// BN fold: s = g * rsqrt(v + 1e-5), t = b - m * s
// -------------------------------------------------------------------------
__global__ void bn_prep(const float* __restrict__ g, const float* __restrict__ b,
                        const float* __restrict__ m, const float* __restrict__ v,
                        float* __restrict__ s, float* __restrict__ t, int n) {
  int i = blockIdx.x * blockDim.x + threadIdx.x;
  if (i < n) {
    float sc = g[i] * rsqrtf(v[i] + 1e-5f);
    s[i] = sc;
    t[i] = b[i] - m[i] * sc;
  }
}

// -------------------------------------------------------------------------
// Weight prepass: WT[n][k] = bf16(W[k][n])  (done once; reused by 391 blocks)
// -------------------------------------------------------------------------
__global__ void wt_bf16(const float* __restrict__ W, bf16* __restrict__ WT,
                        int K, int N) {
  int i = blockIdx.x * blockDim.x + threadIdx.x;
  if (i < K * N) {
    int k = i / N, n = i - k * N;
    WT[(size_t)n * K + k] = (bf16)W[i];
  }
}

// -------------------------------------------------------------------------
// f32 -> bf16 (vectorized), n multiple of 4
// -------------------------------------------------------------------------
__global__ void cvt_bf16(const float* __restrict__ in, bf16* __restrict__ out, int n) {
  int i = (blockIdx.x * blockDim.x + threadIdx.x) * 4;
  if (i < n) {
    float4 v = *(const float4*)&in[i];
    v4bf p;
    p[0] = (bf16)v.x; p[1] = (bf16)v.y; p[2] = (bf16)v.z; p[3] = (bf16)v.w;
    *(v4bf*)&out[i] = p;
  }
}

// -------------------------------------------------------------------------
// out = (1 + *eps) * in   (eps is a device scalar)
// -------------------------------------------------------------------------
__global__ void scale_copy(const float* __restrict__ in, float* __restrict__ out,
                           const float* __restrict__ eps, int n) {
  int i = blockIdx.x * blockDim.x + threadIdx.x;
  float a = 1.0f + *eps;
  if (i < n) out[i] = a * in[i];
}

// -------------------------------------------------------------------------
// GIN scatter: out[dst[e]] += h[src[e]]  (per edge, float4 chunks, f32 atomics)
// -------------------------------------------------------------------------
__global__ void scatter_add(const float* __restrict__ h, const int* __restrict__ src,
                            const int* __restrict__ dst, float* __restrict__ out,
                            int F, int nE) {
  int tid = blockIdx.x * blockDim.x + threadIdx.x;
  int per = F >> 2;
  if (tid >= nE * per) return;
  int e = tid / per;
  int c = (tid - e * per) << 2;
  int sN = src[e], dN = dst[e];
  const float4 v = *(const float4*)&h[(size_t)sN * F + c];
  float* o = &out[(size_t)dN * F + c];
  atomicAdd(o + 0, v.x);
  atomicAdd(o + 1, v.y);
  atomicAdd(o + 2, v.z);
  atomicAdd(o + 3, v.w);
}

// -------------------------------------------------------------------------
// Async copy of 16B from global to LDS (gfx1250 ASYNCcnt path)
// -------------------------------------------------------------------------
__device__ __forceinline__ void async_b128(unsigned lds_off, const bf16* gptr) {
  asm volatile("global_load_async_to_lds_b128 %0, %1, off"
               :: "v"(lds_off), "v"(gptr) : "memory");
}
__device__ __forceinline__ void wait_async0() {
  asm volatile("s_wait_asynccnt 0x0" ::: "memory");
}

// -------------------------------------------------------------------------
// WMMA GEMM: C = epi(A @ WT^T + bias). A: M x K bf16 row-major.
// WT: N x K bf16 (pre-transposed weights). C: f32 or bf16 per OUTBF.
// Block tile 128x128, BK=32, 256 thr = 8 wave32, wave tile 64x32 (4x2 wmma).
// Double-buffered LDS filled by global_load_async_to_lds_b128.
// Epilogue order: +bias -> (relu) -> (y*s + t)
// -------------------------------------------------------------------------
template<int RELU, int BNORM, int OUTBF>
__global__ __launch_bounds__(256)
void wmma_gemm(const bf16* __restrict__ A, const bf16* __restrict__ WT,
               const float* __restrict__ bias, const float* __restrict__ bns,
               const float* __restrict__ bnt, void* __restrict__ Cout,
               int M, int K, int N) {
  constexpr int LDT = 40;                       // 32 k + 8 pad (80B rows, 16B-aligned)
  __shared__ __align__(16) bf16 As[2][128][LDT];
  __shared__ __align__(16) bf16 Bs[2][128][LDT];

  const int tid = threadIdx.x;
  const int m0  = blockIdx.x * 128;
  const int n0  = blockIdx.y * 128;

  // staging assignment: 2 threads per tile row, 2x16B async issues each
  const int r    = tid >> 1;
  const int half = tid & 1;
  int grow = m0 + r; if (grow > M - 1) grow = M - 1;   // clamp; pad rows discarded later
  const bf16* gA = A  + (size_t)grow * K + half * 16;
  const bf16* gB = WT + (size_t)(n0 + r) * K + half * 16;
  const unsigned laBase = (unsigned)(uintptr_t)&As[0][r][half * 16];
  const unsigned lbBase = (unsigned)(uintptr_t)&Bs[0][r][half * 16];
  const unsigned bufStride = (unsigned)(sizeof(bf16) * 128 * LDT);

  auto issue = [&](int kt, int buf) {
    const bf16* pa = gA + kt;
    const bf16* pb = gB + kt;
    unsigned la = laBase + (unsigned)buf * bufStride;
    unsigned lb = lbBase + (unsigned)buf * bufStride;
    async_b128(la,       pa);
    async_b128(la + 16u, pa + 8);
    async_b128(lb,       pb);
    async_b128(lb + 16u, pb + 8);
  };

  const int lane = tid & 31;
  const int wave = tid >> 5;
  const int wm   = wave & 1;
  const int wn   = wave >> 1;
  const int lrow = lane & 15;
  const int lhi  = lane >> 4;

  v8f acc[4][2];
  const v8f vzero = {0.f, 0.f, 0.f, 0.f, 0.f, 0.f, 0.f, 0.f};
  #pragma unroll
  for (int i = 0; i < 4; ++i)
    #pragma unroll
    for (int j = 0; j < 2; ++j) acc[i][j] = vzero;

  const int nk = K >> 5;
  issue(0, 0);

  for (int ik = 0; ik < nk; ++ik) {
    wait_async0();            // my async fills for buf[ik&1] have landed
    __syncthreads();          // everyone's fills visible; prior reads of alt buf done
    if (ik + 1 < nk) issue((ik + 1) << 5, (ik + 1) & 1);   // prefetch next tile

    const int buf = ik & 1;
    V16 af[4], bfr[2];
    const int ak = lhi * 8;   // A-frag K interleave {0-7,16-23} / {8-15,24-31}
    #pragma unroll
    for (int i = 0; i < 4; ++i) {
      const bf16* rp = &As[buf][wm * 64 + i * 16 + lrow][0];
      af[i].h[0] = *(const v8bf*)(rp + ak);
      af[i].h[1] = *(const v8bf*)(rp + ak + 16);
    }
    const int bk = lhi * 8;   // B-frag K interleave {0-7,16-23} / {8-15,24-31}
    #pragma unroll
    for (int j = 0; j < 2; ++j) {
      const bf16* cp = &Bs[buf][wn * 32 + j * 16 + lrow][0];
      bfr[j].h[0] = *(const v8bf*)(cp + bk);
      bfr[j].h[1] = *(const v8bf*)(cp + bk + 16);
    }
    #pragma unroll
    for (int i = 0; i < 4; ++i)
      #pragma unroll
      for (int j = 0; j < 2; ++j)
        acc[i][j] = __builtin_amdgcn_wmma_f32_16x16x32_bf16(
            false, af[i].v, false, bfr[j].v, (short)0, acc[i][j], false, false);
  }

  // ---- epilogue: bias -> relu -> bn-affine; dtype per OUTBF ----
  #pragma unroll
  for (int j = 0; j < 2; ++j) {
    int col  = n0 + wn * 32 + j * 16 + lrow;
    float bb = bias[col];
    float s = 1.f, t = 0.f;
    if (BNORM) { s = bns[col]; t = bnt[col]; }
    #pragma unroll
    for (int i = 0; i < 4; ++i) {
      int rbase = m0 + wm * 64 + i * 16 + lhi * 8;
      #pragma unroll
      for (int rr = 0; rr < 8; ++rr) {
        int row = rbase + rr;
        if (row < M) {
          float v = acc[i][j][rr] + bb;
          if (RELU)  v = fmaxf(v, 0.f);
          if (BNORM) v = v * s + t;
          if (OUTBF) ((bf16*)Cout)[(size_t)row * N + col] = (bf16)v;
          else       ((float*)Cout)[(size_t)row * N + col] = v;
        }
      }
    }
  }
}

// -------------------------------------------------------------------------
// Final head: out[e] = (h[src]*h[dst]) @ fc2_W + fc2_b   (one wave per edge)
// -------------------------------------------------------------------------
__global__ __launch_bounds__(256)
void edge_fc(const float* __restrict__ h, const int* __restrict__ edge_index,
             const int* __restrict__ train_id, const float* __restrict__ fw,
             const float* __restrict__ fb, float* __restrict__ out,
             int nT, int nEdges) {
  int gw   = (blockIdx.x * blockDim.x + threadIdx.x) >> 5;
  int lane = threadIdx.x & 31;
  if (gw >= nT) return;
  int id = train_id[gw];
  int a  = edge_index[id];
  int b  = edge_index[nEdges + id];
  float acc[N_CLASS_C];
  #pragma unroll
  for (int o = 0; o < N_CLASS_C; ++o) acc[o] = 0.f;

  for (int c = lane * 4; c < HIDDEN_C; c += 128) {
    float4 ha = *(const float4*)&h[(size_t)a * HIDDEN_C + c];
    float4 hb = *(const float4*)&h[(size_t)b * HIDDEN_C + c];
    float p[4] = {ha.x * hb.x, ha.y * hb.y, ha.z * hb.z, ha.w * hb.w};
    #pragma unroll
    for (int q = 0; q < 4; ++q) {
      const float* wr = &fw[(size_t)(c + q) * N_CLASS_C];
      #pragma unroll
      for (int o = 0; o < N_CLASS_C; ++o) acc[o] += p[q] * wr[o];
    }
  }
  #pragma unroll
  for (int o = 0; o < N_CLASS_C; ++o)
    #pragma unroll
    for (int off = 16; off > 0; off >>= 1)
      acc[o] += __shfl_xor(acc[o], off, 32);
  if (lane < N_CLASS_C) out[(size_t)gw * N_CLASS_C + lane] = acc[lane] + fb[lane];
}

// -------------------------------------------------------------------------
extern "C" void kernel_launch(void* const* d_in, const int* in_sizes, int n_in,
                              void* d_out, int out_size, void* d_ws, size_t ws_size,
                              hipStream_t stream) {
  const float* x          = (const float*)d_in[0];
  const int*   edge_index = (const int*)  d_in[1];
  const int*   train_id   = (const int*)  d_in[2];
  const float* W1 = (const float*)d_in[3];  const float* b1 = (const float*)d_in[4];
  const float* W2 = (const float*)d_in[5];  const float* b2 = (const float*)d_in[6];
  const float* W3 = (const float*)d_in[7];  const float* b3 = (const float*)d_in[8];
  const float* bn1_g = (const float*)d_in[9];
  const float* bn1_b = (const float*)d_in[10];
  const float* bn1_m = (const float*)d_in[11];
  const float* bn1_v = (const float*)d_in[12];
  const float* eps1  = (const float*)d_in[13];
  const float* W4 = (const float*)d_in[14]; const float* b4 = (const float*)d_in[15];
  const float* bn2_g = (const float*)d_in[16];
  const float* bn2_b = (const float*)d_in[17];
  const float* bn2_m = (const float*)d_in[18];
  const float* bn2_v = (const float*)d_in[19];
  const float* eps2  = (const float*)d_in[20];
  const float* lin1W = (const float*)d_in[21]; const float* lin1b = (const float*)d_in[22];
  const float* lin2W = (const float*)d_in[23]; const float* lin2b = (const float*)d_in[24];
  const float* fcW   = (const float*)d_in[25]; const float* fcb   = (const float*)d_in[26];

  // ---- workspace layout ----
  char*  ws = (char*)d_ws;
  size_t off = 0;
  float* s1 = (float*)(ws + off);  float* t1 = s1 + 512;
  float* s2 = t1 + 512;            float* t2 = s2 + 512;
  off = 8192;
  const size_t WSMALL = (size_t)IN_F_C * HIDDEN_C * sizeof(bf16);
  const size_t WBIG   = (size_t)HIDDEN_C * HIDDEN_C * sizeof(bf16);
  bf16* W1t = (bf16*)(ws + off); off += WSMALL;
  bf16* W2t = (bf16*)(ws + off); off += WBIG;
  bf16* W3t = (bf16*)(ws + off); off += WBIG;
  bf16* W4t = (bf16*)(ws + off); off += WBIG;
  bf16* L1t = (bf16*)(ws + off); off += WBIG;
  bf16* L2t = (bf16*)(ws + off); off += WBIG;
  off = (off + 255) & ~(size_t)255;
  const size_t HB = (size_t)N_NODES_C * HIDDEN_C * sizeof(bf16);
  const size_t HF = (size_t)N_NODES_C * HIDDEN_C * sizeof(float);
  bf16*  bB0 = (bf16*)(ws + off);  off += HB;
  bf16*  bB1 = (bf16*)(ws + off);  off += HB;
  float* bF0 = (float*)(ws + off); off += HF;
  float* bF1 = (float*)(ws + off); off += HF;

  // ---- prepasses: BN fold + weight transpose/convert ----
  bn_prep<<<2, 256, 0, stream>>>(bn1_g, bn1_b, bn1_m, bn1_v, s1, t1, HIDDEN_C);
  bn_prep<<<2, 256, 0, stream>>>(bn2_g, bn2_b, bn2_m, bn2_v, s2, t2, HIDDEN_C);
  wt_bf16<<<(IN_F_C * HIDDEN_C + 255) / 256, 256, 0, stream>>>(W1, W1t, IN_F_C, HIDDEN_C);
  wt_bf16<<<(HIDDEN_C * HIDDEN_C + 255) / 256, 256, 0, stream>>>(W2, W2t, HIDDEN_C, HIDDEN_C);
  wt_bf16<<<(HIDDEN_C * HIDDEN_C + 255) / 256, 256, 0, stream>>>(W3, W3t, HIDDEN_C, HIDDEN_C);
  wt_bf16<<<(HIDDEN_C * HIDDEN_C + 255) / 256, 256, 0, stream>>>(W4, W4t, HIDDEN_C, HIDDEN_C);
  wt_bf16<<<(HIDDEN_C * HIDDEN_C + 255) / 256, 256, 0, stream>>>(lin1W, L1t, HIDDEN_C, HIDDEN_C);
  wt_bf16<<<(HIDDEN_C * HIDDEN_C + 255) / 256, 256, 0, stream>>>(lin2W, L2t, HIDDEN_C, HIDDEN_C);

  // ---- GIN aggregation #1 (F=64, f32): bF0 = (1+eps1)*x + scatter(x); -> bf16 ----
  int n0 = N_NODES_C * IN_F_C;
  scale_copy<<<(n0 + 255) / 256, 256, 0, stream>>>(x, bF0, eps1, n0);
  int na1 = N_EDGES_C * (IN_F_C / 4);
  scatter_add<<<(na1 + 255) / 256, 256, 0, stream>>>(
      x, edge_index, edge_index + N_EDGES_C, bF0, IN_F_C, N_EDGES_C);
  cvt_bf16<<<(n0 / 4 + 255) / 256, 256, 0, stream>>>(bF0, bB0, n0);

  dim3 gg((N_NODES_C + 127) / 128, HIDDEN_C / 128);
  wmma_gemm<1, 0, 1><<<gg, 256, 0, stream>>>(bB0, W1t, b1, nullptr, nullptr, bB1,
                                             N_NODES_C, IN_F_C, HIDDEN_C);
  wmma_gemm<1, 0, 1><<<gg, 256, 0, stream>>>(bB1, W2t, b2, nullptr, nullptr, bB0,
                                             N_NODES_C, HIDDEN_C, HIDDEN_C);
  wmma_gemm<1, 1, 0><<<gg, 256, 0, stream>>>(bB0, W3t, b3, s1, t1, bF0,
                                             N_NODES_C, HIDDEN_C, HIDDEN_C);

  // ---- GIN aggregation #2 (F=512, f32): bF1 = (1+eps2)*bF0 + scatter(bF0) ----
  int n1 = N_NODES_C * HIDDEN_C;
  scale_copy<<<(n1 + 255) / 256, 256, 0, stream>>>(bF0, bF1, eps2, n1);
  int na2 = N_EDGES_C * (HIDDEN_C / 4);
  scatter_add<<<(na2 + 255) / 256, 256, 0, stream>>>(
      bF0, edge_index, edge_index + N_EDGES_C, bF1, HIDDEN_C, N_EDGES_C);
  cvt_bf16<<<(n1 / 4 + 255) / 256, 256, 0, stream>>>(bF1, bB0, n1);

  wmma_gemm<1, 1, 1><<<gg, 256, 0, stream>>>(bB0, W4t, b4, s2, t2, bB1,
                                             N_NODES_C, HIDDEN_C, HIDDEN_C);
  wmma_gemm<1, 0, 1><<<gg, 256, 0, stream>>>(bB1, L1t, lin1b, nullptr, nullptr, bB0,
                                             N_NODES_C, HIDDEN_C, HIDDEN_C);
  wmma_gemm<0, 0, 0><<<gg, 256, 0, stream>>>(bB0, L2t, lin2b, nullptr, nullptr, bF0,
                                             N_NODES_C, HIDDEN_C, HIDDEN_C);

  edge_fc<<<(N_TRAIN_C * 32 + 255) / 256, 256, 0, stream>>>(
      bF0, edge_index, train_id, fcW, fcb, (float*)d_out, N_TRAIN_C, N_EDGES_C);
}
